// FullSelfAttention_65335042507270
// MI455X (gfx1250) — matrix operand order repaired and checked by reading, not compiled, not run on hardware
//
#include <hip/hip_runtime.h>
#include <hip/hip_bf16.h>
#include <math.h>

typedef unsigned short bf16_t;
typedef __attribute__((ext_vector_type(16))) __bf16 v16bf;
typedef __attribute__((ext_vector_type(8)))  float  v8f;

#define D_MODEL 512
#define N_HEADS 8
#define HEAD_DIM 64
#define SEQ_LEN 4096
#define BATCH 2
#define N_TOK (BATCH * SEQ_LEN)   // 8192

__device__ __forceinline__ bf16_t f2bf(float f){
  unsigned u = __float_as_uint(f);
  u += 0x7FFFu + ((u >> 16) & 1u);     // round-to-nearest-even
  return (bf16_t)(u >> 16);
}

union ABFrag { int i[8]; v16bf v; };

__device__ __forceinline__ v8f wmma_bf16(const int* a, const int* b, v8f c){
  ABFrag ua, ub;
#pragma unroll
  for (int j = 0; j < 8; ++j) { ua.i[j] = a[j]; ub.i[j] = b[j]; }
  return __builtin_amdgcn_wmma_f32_16x16x32_bf16(false, ua.v, false, ub.v,
                                                 (short)0, c, false, false);
}

// A fragment (16x32 bf16): row m = m0 + (lane&15); lanes 16-31 take K+8.
__device__ __forceinline__ void load_a_frag(const bf16_t* A, int lda, int m0, int k0,
                                            int lane, int* out){
  const int half8 = (lane >> 4) << 3;
  const bf16_t* p = A + (m0 + (lane & 15)) * lda + k0 + half8;
#pragma unroll
  for (int j = 0; j < 8; ++j){
    int kd = ((j >> 2) << 4) + ((j & 3) << 1);
    out[j] = *(const int*)(p + kd);
  }
}

// B fragment (32x16 bf16): col n = n0 + (lane&15); lanes 16-31 take K+8.
// Source stored K-contiguous: element (k, n) at n*ldb + k.
__device__ __forceinline__ void load_b_frag(const bf16_t* Bt, int ldb, int n0, int k0,
                                            int lane, int* out){
  const int half16 = (lane >> 4) << 3;
  const bf16_t* p = Bt + (n0 + (lane & 15)) * ldb + k0 + half16;
#pragma unroll
  for (int j = 0; j < 8; ++j) out[j] = *(const int*)(p + ((j >> 2) << 4) + ((j & 3) << 1));
}

// One wave computes a 32x64 strip (2x4 accumulator tiles), double-buffered frags:
// next K-step loads are issued before the current K-step WMMAs so loads overlap XDL.
__device__ __forceinline__ void gemm_mainloop(const bf16_t* A, const bf16_t* Bt, int K,
                                              int m0, int n0, int lane, v8f acc[2][4]){
  int a[2][2][8];     // [buf][mi][frag]
  int b[2][4][8];     // [buf][ni][frag]
#pragma unroll
  for (int mi = 0; mi < 2; ++mi) load_a_frag(A, K, m0 + mi * 16, 0, lane, a[0][mi]);
#pragma unroll
  for (int ni = 0; ni < 4; ++ni) load_b_frag(Bt, K, n0 + ni * 16, 0, lane, b[0][ni]);

  for (int k0 = 0; k0 < K; k0 += 32){
    const int cur = (k0 >> 5) & 1, nxt = cur ^ 1;
    if (k0 + 32 < K){
#pragma unroll
      for (int mi = 0; mi < 2; ++mi)
        load_a_frag(A, K, m0 + mi * 16, k0 + 32, lane, a[nxt][mi]);
#pragma unroll
      for (int ni = 0; ni < 4; ++ni)
        load_b_frag(Bt, K, n0 + ni * 16, k0 + 32, lane, b[nxt][ni]);
    }
#pragma unroll
    for (int mi = 0; mi < 2; ++mi)
#pragma unroll
      for (int ni = 0; ni < 4; ++ni)
        acc[mi][ni] = wmma_bf16(a[cur][mi], b[cur][ni], acc[mi][ni]);
  }
}

// ---------------- Stage 1: f32 -> bf16 conversions ----------------
__global__ void cvt_kernel(const float* __restrict__ in, bf16_t* __restrict__ out, int n){
  int i = (blockIdx.x * blockDim.x + threadIdx.x) * 8;
  if (i < n){
    float4 f0 = *(const float4*)(in + i);
    float4 f1 = *(const float4*)(in + i + 4);
    uint4 d;
    d.x = (unsigned)f2bf(f0.x) | ((unsigned)f2bf(f0.y) << 16);
    d.y = (unsigned)f2bf(f0.z) | ((unsigned)f2bf(f0.w) << 16);
    d.z = (unsigned)f2bf(f1.x) | ((unsigned)f2bf(f1.y) << 16);
    d.w = (unsigned)f2bf(f1.z) | ((unsigned)f2bf(f1.w) << 16);
    *(uint4*)(out + i) = d;
  }
}

// in [R,C] row-major -> out [C,R] (so B frags read K-contiguous)
__global__ void cvt_transpose_kernel(const float* __restrict__ in, bf16_t* __restrict__ out,
                                     int R, int C){
  int i = blockIdx.x * blockDim.x + threadIdx.x;
  if (i < R * C){
    int r = i / C, c = i % C;
    out[c * R + r] = f2bf(in[i]);
  }
}

// ---------------- Stage 2: QKV projection GEMM ----------------
__global__ __launch_bounds__(128) void qkv_gemm_kernel(
    const bf16_t* __restrict__ xb, const bf16_t* __restrict__ wqkvT,
    const float* __restrict__ b_qkv,
    bf16_t* __restrict__ q, bf16_t* __restrict__ k, bf16_t* __restrict__ vT){
  const int lane = threadIdx.x & 31;
  const int wave = threadIdx.x >> 5;
  const int half = lane >> 4;
  const int m0 = blockIdx.x * 128 + wave * 32;
  const int n0 = blockIdx.y * 64;

  v8f acc[2][4] = {};
  gemm_mainloop(xb, wqkvT, D_MODEL, m0, n0, lane, acc);

#pragma unroll
  for (int mi = 0; mi < 2; ++mi){
#pragma unroll
    for (int s = 0; s < 4; ++s){
      int n = n0 + s * 16 + (lane & 15);
      float bias = b_qkv[n];
      int part = n >> 9;          // 0:q 1:k 2:v
      int nn = n & 511;
      int h = nn >> 6;
      int e = nn & 63;
#pragma unroll
      for (int r = 0; r < 8; ++r){
        int mg = m0 + mi * 16 + r + (half << 3);
        int bb = mg >> 12;
        int l  = mg & (SEQ_LEN - 1);
        int bh = bb * N_HEADS + h;
        float v = acc[mi][s][r] + bias;
        if (part == 0)      q[(bh * SEQ_LEN + l) * HEAD_DIM + e] = f2bf(v * 0.125f); // fold 1/sqrt(d)
        else if (part == 1) k[(bh * SEQ_LEN + l) * HEAD_DIM + e] = f2bf(v);
        else                vT[(bh * HEAD_DIM + e) * SEQ_LEN + l] = f2bf(v);         // transposed V
      }
    }
  }
}

// ---------------- Stage 3: flash attention ----------------
__global__ __launch_bounds__(128) void flash_attn_kernel(
    const bf16_t* __restrict__ q, const bf16_t* __restrict__ k,
    const bf16_t* __restrict__ vT, bf16_t* __restrict__ ao){
  const int lane = threadIdx.x & 31;
  const int wave = threadIdx.x >> 5;
  const int half = lane >> 4;
  const int b = blockIdx.z, h = blockIdx.y;
  const int l0 = blockIdx.x * 64 + wave * 16;
  const int bh = b * N_HEADS + h;
  const bf16_t* qh = q  + bh * SEQ_LEN * HEAD_DIM;
  const bf16_t* kh = k  + bh * SEQ_LEN * HEAD_DIM;
  const bf16_t* vh = vT + bh * HEAD_DIM * SEQ_LEN;

  __shared__ bf16_t ldsP[4][16][68];   // wave-private P tiles (same-wave DS is in-order)

  int aq[2][8];
  load_a_frag(qh, HEAD_DIM, l0, 0,  lane, aq[0]);
  load_a_frag(qh, HEAD_DIM, l0, 32, lane, aq[1]);

  v8f o[4] = {};
  float mrow[8], lrow[8];
#pragma unroll
  for (int r = 0; r < 8; ++r){ mrow[r] = -1e30f; lrow[r] = 0.f; }

  for (int s0 = 0; s0 < SEQ_LEN; s0 += 64){
    // prefetch next K/V tiles into cache (global_prefetch_b8) while we compute
    if (s0 + 64 < SEQ_LEN){
      __builtin_prefetch(kh + (s0 + 64 + lane) * HEAD_DIM, 0, 1);
      __builtin_prefetch(kh + (s0 + 96 + lane) * HEAD_DIM, 0, 1);
      __builtin_prefetch(vh + lane        * SEQ_LEN + s0 + 64, 0, 1);
      __builtin_prefetch(vh + (lane + 32) * SEQ_LEN + s0 + 64, 0, 1);
    }

    // S = Q K^T : load all 8 K B-frags as one clause, then 8 WMMAs
    int bk0[4][8], bk1[4][8];
#pragma unroll
    for (int s = 0; s < 4; ++s){
      load_b_frag(kh, HEAD_DIM, s0 + s * 16, 0,  lane, bk0[s]);
      load_b_frag(kh, HEAD_DIM, s0 + s * 16, 32, lane, bk1[s]);
    }
    v8f sc[4] = {};
#pragma unroll
    for (int s = 0; s < 4; ++s){
      sc[s] = wmma_bf16(aq[0], bk0[s], sc[s]);
      sc[s] = wmma_bf16(aq[1], bk1[s], sc[s]);
    }

    // issue V B-frag loads now; latency hides under softmax below
    int bv[2][4][8];
#pragma unroll
    for (int kk = 0; kk < 2; ++kk)
#pragma unroll
      for (int s = 0; s < 4; ++s)
        load_b_frag(vh, SEQ_LEN, s * 16, s0 + kk * 32, lane, bv[kk][s]);

    // online softmax (scale already folded into Q): row r stats per half-wave
#pragma unroll
    for (int r = 0; r < 8; ++r){
      float x0 = sc[0][r], x1 = sc[1][r], x2 = sc[2][r], x3 = sc[3][r];
      float loc = fmaxf(fmaxf(x0, x1), fmaxf(x2, x3));
      loc = fmaxf(loc, __shfl_xor(loc, 1));
      loc = fmaxf(loc, __shfl_xor(loc, 2));
      loc = fmaxf(loc, __shfl_xor(loc, 4));
      loc = fmaxf(loc, __shfl_xor(loc, 8));
      float mnew  = fmaxf(mrow[r], loc);
      float alpha = __expf(mrow[r] - mnew);
      mrow[r] = mnew;
      lrow[r] *= alpha;
#pragma unroll
      for (int s = 0; s < 4; ++s) o[s][r] *= alpha;
      float p0 = __expf(x0 - mnew), p1 = __expf(x1 - mnew);
      float p2 = __expf(x2 - mnew), p3 = __expf(x3 - mnew);
      sc[0][r] = p0; sc[1][r] = p1; sc[2][r] = p2; sc[3][r] = p3;
      float rs = (p0 + p1) + (p2 + p3);
      rs += __shfl_xor(rs, 1); rs += __shfl_xor(rs, 2);
      rs += __shfl_xor(rs, 4); rs += __shfl_xor(rs, 8);
      lrow[r] += rs;
    }

    // spill P (C layout) to LDS as bf16
#pragma unroll
    for (int s = 0; s < 4; ++s){
#pragma unroll
      for (int r = 0; r < 8; ++r)
        ldsP[wave][r + (half << 3)][s * 16 + (lane & 15)] = f2bf(sc[s][r]);
    }

    // O += P V  (A frags re-read from LDS in A layout)
#pragma unroll
    for (int kk = 0; kk < 2; ++kk){
      int ap[8];
      const bf16_t* prow = &ldsP[wave][lane & 15][kk * 32 + (half << 3)];
#pragma unroll
      for (int j = 0; j < 8; ++j){
        int kd = ((j >> 2) << 4) + ((j & 3) << 1);
        ap[j] = *(const int*)(prow + kd);
      }
#pragma unroll
      for (int s = 0; s < 4; ++s)
        o[s] = wmma_bf16(ap, bv[kk][s], o[s]);
    }
  }

  // epilogue: normalize, write attention output [B, L, D] bf16
  float invl[8];
#pragma unroll
  for (int r = 0; r < 8; ++r) invl[r] = 1.0f / lrow[r];
#pragma unroll
  for (int s = 0; s < 4; ++s){
    int col = h * HEAD_DIM + s * 16 + (lane & 15);
#pragma unroll
    for (int r = 0; r < 8; ++r){
      int tok = b * SEQ_LEN + l0 + r + (half << 3);
      ao[tok * D_MODEL + col] = f2bf(o[s][r] * invl[r]);
    }
  }
}

// ---------------- Stage 4: output projection GEMM (f32 out) ----------------
__global__ __launch_bounds__(128) void out_gemm_kernel(
    const bf16_t* __restrict__ ao, const bf16_t* __restrict__ woutT,
    const float* __restrict__ b_out, float* __restrict__ out){
  const int lane = threadIdx.x & 31;
  const int wave = threadIdx.x >> 5;
  const int half = lane >> 4;
  const int m0 = blockIdx.x * 128 + wave * 32;
  const int n0 = blockIdx.y * 64;

  v8f acc[2][4] = {};
  gemm_mainloop(ao, woutT, D_MODEL, m0, n0, lane, acc);

#pragma unroll
  for (int mi = 0; mi < 2; ++mi){
#pragma unroll
    for (int s = 0; s < 4; ++s){
      int n = n0 + s * 16 + (lane & 15);
      float bias = b_out[n];
#pragma unroll
      for (int r = 0; r < 8; ++r){
        int mg = m0 + mi * 16 + r + (half << 3);
        out[mg * D_MODEL + n] = acc[mi][s][r] + bias;
      }
    }
  }
}

extern "C" void kernel_launch(void* const* d_in, const int* in_sizes, int n_in,
                              void* d_out, int out_size, void* d_ws, size_t ws_size,
                              hipStream_t stream){
  (void)in_sizes; (void)n_in; (void)out_size; (void)ws_size;
  const float* x     = (const float*)d_in[0];
  const float* W_qkv = (const float*)d_in[1];
  const float* b_qkv = (const float*)d_in[2];
  const float* W_out = (const float*)d_in[3];
  const float* b_out = (const float*)d_in[4];
  float* out = (float*)d_out;

  char* ws = (char*)d_ws;
  bf16_t* xb    = (bf16_t*)(ws + 0);          // 8192*512   bf16 =  8 MB
  bf16_t* wqkvT = (bf16_t*)(ws + 8388608);    // 1536*512   bf16 = 1.5 MB
  bf16_t* woutT = (bf16_t*)(ws + 9961472);    //  512*512   bf16 = 0.5 MB
  bf16_t* q     = (bf16_t*)(ws + 10485760);   // [B,H,L,d]  bf16 =  8 MB (pre-scaled)
  bf16_t* k     = (bf16_t*)(ws + 18874368);   // [B,H,L,d]  bf16 =  8 MB
  bf16_t* vT    = (bf16_t*)(ws + 27262976);   // [B,H,d,L]  bf16 =  8 MB
  bf16_t* ao    = (bf16_t*)(ws + 35651584);   // [B,L,D]    bf16 =  8 MB

  cvt_kernel<<<(N_TOK * D_MODEL / 8 + 255) / 256, 256, 0, stream>>>(x, xb, N_TOK * D_MODEL);
  cvt_transpose_kernel<<<(D_MODEL * 3 * D_MODEL + 255) / 256, 256, 0, stream>>>(
      W_qkv, wqkvT, D_MODEL, 3 * D_MODEL);
  cvt_transpose_kernel<<<(D_MODEL * D_MODEL + 255) / 256, 256, 0, stream>>>(
      W_out, woutT, D_MODEL, D_MODEL);

  qkv_gemm_kernel<<<dim3(N_TOK / 128, (3 * D_MODEL) / 64, 1), 128, 0, stream>>>(
      xb, wqkvT, b_qkv, q, k, vT);
  flash_attn_kernel<<<dim3(SEQ_LEN / 64, N_HEADS, BATCH), 128, 0, stream>>>(
      q, k, vT, ao);
  out_gemm_kernel<<<dim3(N_TOK / 128, D_MODEL / 64, 1), 128, 0, stream>>>(
      ao, woutT, b_out, out);
}
